// RelPositionMultiHeadAttention_53068615910311
// MI455X (gfx1250) — matrix operand order repaired and checked by reading, not compiled, not run on hardware
//
#include <hip/hip_runtime.h>

// ---------------------------------------------------------------------------
// CDNA5 (gfx1250) wave32 WMMA helpers
// ---------------------------------------------------------------------------
typedef __attribute__((ext_vector_type(16))) __bf16 v16bf;
typedef __attribute__((ext_vector_type(8)))  float  v8f;

struct alignas(16) U4 { unsigned int x, y, z, w; };
struct alignas(16) F4 { float x, y, z, w; };

union ABFrag { v16bf v; U4 q[2]; __bf16 h[16]; };
union Acc8   { v8f   v; float f[8]; };

__device__ __forceinline__ v8f wmma_bf16(v16bf a, v16bf b, v8f c) {
  // D = A(16x32 bf16) * B(32x16 bf16) + C(16x16 f32)
  return __builtin_amdgcn_wmma_f32_16x16x32_bf16(false, a, false, b,
                                                 (short)0, c, false, false);
}
__device__ __forceinline__ __bf16 to_bf16(float x) { return (__bf16)x; }

// ---------------------------------------------------------------------------
// Generic GEMM: out = X(f32 MxK) @ W(f32 KxN) + bias, bf16/f32/transposed out.
// Block tile 128x64, 8 waves (4M x 2N), each wave 32x32 (2x2 WMMA register
// block -> 4 v_wmma per K-step), K-step 32.
// mode 0: bf16 row-major out0 (+bias+add0); optional out1 (+bias+add1)
// mode 1: f32 row-major out0
// mode 2: bf16 transposed (b,h,dk,t) out0   (rows=(b,t), cols=(h,dk))
// ---------------------------------------------------------------------------
#define LDSTR 48  // padded bf16 row stride (96B, 16B-aligned rows)

__global__ __launch_bounds__(256) void gemm_bf16_wmma(
    const float* __restrict__ X, const float* __restrict__ W,
    const float* __restrict__ bias, const float* __restrict__ add0,
    const float* __restrict__ add1, void* __restrict__ out0,
    void* __restrict__ out1, int M, int N, int K, int mode, int Tdim)
{
  __shared__ __bf16 As[128 * LDSTR];  // [m][k]
  __shared__ __bf16 Bs[64 * LDSTR];   // transposed: [n][k]
  const int tid  = threadIdx.x;
  const int lane = tid & 31, wid = tid >> 5;
  const int wm = wid >> 1, wn = wid & 1;    // 4 M-waves x 2 N-waves
  const int half = lane >> 4, lan = lane & 15;
  const int row0 = blockIdx.x * 128;
  const int col0 = blockIdx.y * 64;

  Acc8 acc[2][2];  // [sm][sn]
#pragma unroll
  for (int sm = 0; sm < 2; ++sm)
#pragma unroll
    for (int sn = 0; sn < 2; ++sn)
#pragma unroll
      for (int i = 0; i < 8; ++i) acc[sm][sn].f[i] = 0.0f;

  const int ar  = tid >> 1;          // 0..127 (A row)
  const int ac_ = (tid & 1) * 16;    // 0/16   (A col base, 16 floats each)
  const int bk  = tid >> 3;          // 0..31  (B k)
  const int bn  = (tid & 7) * 8;     // B n base

  for (int k0 = 0; k0 < K; k0 += 32) {
    {  // stage A tile (128x32), f32 -> bf16
      F4 x0 = {0, 0, 0, 0}, x1 = {0, 0, 0, 0}, x2 = {0, 0, 0, 0},
         x3 = {0, 0, 0, 0};
      const int gr = row0 + ar;
      if (gr < M) {
        const float* src = X + (size_t)gr * K + k0 + ac_;
        x0 = *(const F4*)src;
        x1 = *(const F4*)(src + 4);
        x2 = *(const F4*)(src + 8);
        x3 = *(const F4*)(src + 12);
        if (k0 + 32 < K) __builtin_prefetch(src + 32, 0, 1);
      }
      __bf16* d = &As[ar * LDSTR + ac_];
      d[0]  = to_bf16(x0.x); d[1]  = to_bf16(x0.y);
      d[2]  = to_bf16(x0.z); d[3]  = to_bf16(x0.w);
      d[4]  = to_bf16(x1.x); d[5]  = to_bf16(x1.y);
      d[6]  = to_bf16(x1.z); d[7]  = to_bf16(x1.w);
      d[8]  = to_bf16(x2.x); d[9]  = to_bf16(x2.y);
      d[10] = to_bf16(x2.z); d[11] = to_bf16(x2.w);
      d[12] = to_bf16(x3.x); d[13] = to_bf16(x3.y);
      d[14] = to_bf16(x3.z); d[15] = to_bf16(x3.w);
    }
    {  // stage B tile (32x64) transposed -> Bs[n][k]
      const float* src = W + (size_t)(k0 + bk) * N + col0 + bn;
      F4 w0 = *(const F4*)src;
      F4 w1 = *(const F4*)(src + 4);
      if (k0 + 32 < K) __builtin_prefetch(src + (size_t)32 * N, 0, 1);
      Bs[(bn + 0) * LDSTR + bk] = to_bf16(w0.x);
      Bs[(bn + 1) * LDSTR + bk] = to_bf16(w0.y);
      Bs[(bn + 2) * LDSTR + bk] = to_bf16(w0.z);
      Bs[(bn + 3) * LDSTR + bk] = to_bf16(w0.w);
      Bs[(bn + 4) * LDSTR + bk] = to_bf16(w1.x);
      Bs[(bn + 5) * LDSTR + bk] = to_bf16(w1.y);
      Bs[(bn + 6) * LDSTR + bk] = to_bf16(w1.z);
      Bs[(bn + 7) * LDSTR + bk] = to_bf16(w1.w);
    }
    __syncthreads();
    // A-fragments: lane m=lan; K = 16*half + e
    ABFrag af[2];
#pragma unroll
    for (int sm = 0; sm < 2; ++sm) {
      const __bf16* arow = &As[(wm * 32 + sm * 16 + lan) * LDSTR];
      af[sm].q[0] = *(const U4*)(arow + half * 16);
      af[sm].q[1] = *(const U4*)(arow + half * 16 + 8);
    }
    // B-fragments: lane n=lan; K = 16*half + e
    ABFrag bfg[2];
#pragma unroll
    for (int sn = 0; sn < 2; ++sn) {
      const __bf16* brow = &Bs[(wn * 32 + sn * 16 + lan) * LDSTR];
      bfg[sn].q[0] = *(const U4*)(brow + half * 16);
      bfg[sn].q[1] = *(const U4*)(brow + half * 16 + 8);
    }
#pragma unroll
    for (int sm = 0; sm < 2; ++sm)
#pragma unroll
      for (int sn = 0; sn < 2; ++sn)
        acc[sm][sn].v = wmma_bf16(af[sm].v, bfg[sn].v, acc[sm][sn].v);
    __syncthreads();
  }

  // epilogue: D layout M = a + 8*half, N = lan
#pragma unroll
  for (int sn = 0; sn < 2; ++sn) {
    const int n = col0 + wn * 32 + sn * 16 + lan;
    const float bv = bias ? bias[n] : 0.0f;
    const float a0 = add0 ? add0[n] : 0.0f;
    const float a1 = add1 ? add1[n] : 0.0f;
#pragma unroll
    for (int sm = 0; sm < 2; ++sm) {
#pragma unroll
      for (int a = 0; a < 8; ++a) {
        const int m  = wm * 32 + sm * 16 + a + half * 8;
        const int gr = row0 + m;
        if (gr >= M) continue;
        const float val = acc[sm][sn].f[a] + bv;
        if (mode == 0) {
          ((__bf16*)out0)[(size_t)gr * N + n] = to_bf16(val + a0);
          if (out1) ((__bf16*)out1)[(size_t)gr * N + n] = to_bf16(val + a1);
        } else if (mode == 1) {
          ((float*)out0)[(size_t)gr * N + n] = val;
        } else {  // V transposed: (b, h, dk, t)
          const int b = gr / Tdim, t = gr % Tdim;
          const int hh = n >> 6, dk = n & 63;
          ((__bf16*)out0)[(((size_t)(b * 16 + hh) * 64 + dk) * Tdim) + t] =
              to_bf16(val);
        }
      }
    }
  }
}

// ---------------------------------------------------------------------------
// Fused rel-pos flash attention. One wave per (b, h, 16-query tile).
// bd_shifted[i,j] = qv[i] . p[j - i + T - 1]  (rel_shift fused via G0/G1 tiles
// + cross-lane diagonal select).
// ---------------------------------------------------------------------------
__global__ __launch_bounds__(256) void relattn_wmma(
    const __bf16* __restrict__ qu, const __bf16* __restrict__ qv,
    const __bf16* __restrict__ kb, const __bf16* __restrict__ vT,
    const __bf16* __restrict__ pb, const int* __restrict__ mask,
    float* __restrict__ xout)
{
  constexpr int T = 1024, H = 16, DK = 64, F = 1024, P = 2047;
  __shared__ __bf16 Pl[8 * 16 * 32];  // per-wave 16x32 bf16 P staging
  const int tid = threadIdx.x, lane = tid & 31, wid = tid >> 5;
  const int half = lane >> 4, lan = lane & 15;
  __bf16* myP = &Pl[wid * (16 * 32)];
  const int w  = blockIdx.x * 8 + wid;     // 0..2047
  const int it = w & 63, h = (w >> 6) & 15, b = w >> 10;
  const int i0 = it * 16;

  // q_u / q_v A-fragments, K = dk 0..63 (2 frags each)
  ABFrag quf[2], qvf[2];
  {
    const __bf16* qrow = qu + ((size_t)(b * T + i0 + lan) * F) + h * 64;
    const __bf16* vrow = qv + ((size_t)(b * T + i0 + lan) * F) + h * 64;
#pragma unroll
    for (int kk = 0; kk < 2; ++kk) {
      quf[kk].q[0] = *(const U4*)(qrow + kk * 32 + half * 16);
      quf[kk].q[1] = *(const U4*)(qrow + kk * 32 + half * 16 + 8);
      qvf[kk].q[0] = *(const U4*)(vrow + kk * 32 + half * 16);
      qvf[kk].q[1] = *(const U4*)(vrow + kk * 32 + half * 16 + 8);
    }
  }

  Acc8 O[4];
#pragma unroll
  for (int nt = 0; nt < 4; ++nt)
#pragma unroll
    for (int a = 0; a < 8; ++a) O[nt].f[a] = 0.0f;
  float mi[8], li[8];
#pragma unroll
  for (int a = 0; a < 8; ++a) { mi[a] = -1e30f; li[a] = 0.0f; }

  for (int j0 = 0; j0 < T; j0 += 32) {
    Acc8 S[2];
#pragma unroll
    for (int s = 0; s < 2; ++s) {
      const int jb = j0 + s * 16;
      // --- ac = q_u . k^T : K-tile B-fragments (dk 0..31, 32..63) ---
      ABFrag kf0, kf1;
      const __bf16* krow = kb + ((size_t)(b * T + jb + lan) * F) + h * 64;
      kf0.q[0] = *(const U4*)(krow + half * 16);
      kf0.q[1] = *(const U4*)(krow + half * 16 + 8);
      kf1.q[0] = *(const U4*)(krow + 32 + half * 16);
      kf1.q[1] = *(const U4*)(krow + 32 + half * 16 + 8);
      Acc8 ac;
      v8f z0 = {0, 0, 0, 0, 0, 0, 0, 0};
      ac.v = wmma_bf16(quf[0].v, kf0.v, z0);
      ac.v = wmma_bf16(quf[1].v, kf1.v, ac.v);
      // --- bd: target p-row = pbase + (c + 15 - m), pbase = jb-i0+T-16 ---
      const int pbase = jb - i0 + (T - 1 - 15);
      Acc8 G[2];
#pragma unroll
      for (int g = 0; g < 2; ++g) {
        const int pr = pbase + g * 16 + lan;
        ABFrag pf0, pf1;
        if (pr < P) {
          const __bf16* prow = pb + (size_t)pr * F + h * 64;
          pf0.q[0] = *(const U4*)(prow + half * 16);
          pf0.q[1] = *(const U4*)(prow + half * 16 + 8);
          pf1.q[0] = *(const U4*)(prow + 32 + half * 16);
          pf1.q[1] = *(const U4*)(prow + 32 + half * 16 + 8);
        } else {
          U4 zz = {0, 0, 0, 0};
          pf0.q[0] = zz; pf0.q[1] = zz; pf1.q[0] = zz; pf1.q[1] = zz;
        }
        v8f zg = {0, 0, 0, 0, 0, 0, 0, 0};
        G[g].v = wmma_bf16(qvf[0].v, pf0.v, zg);
        G[g].v = wmma_bf16(qvf[1].v, pf1.v, G[g].v);
      }
      // --- diagonal select (rel_shift), mask, scale ---
#pragma unroll
      for (int a = 0; a < 8; ++a) {
        const int m   = a + half * 8;
        const int cpm = lan + 15 - m;              // in [0,30]
        const int src = (half << 4) | (cpm & 15);  // same register index a
        const float g0 = __shfl(G[0].f[a], src, 32);
        const float g1 = __shfl(G[1].f[a], src, 32);
        const float bd = (cpm < 16) ? g0 : g1;
        float sc = (ac.f[a] + bd) * 0.125f;        // 1/sqrt(64)
        const int mv = mask[((size_t)(b * T + i0 + m)) * T + jb + lan];
        sc = (mv > 0) ? sc : -10000.0f;
        S[s].f[a] = sc;
      }
    }
    // --- online softmax over this 32-column strip ---
#pragma unroll
    for (int a = 0; a < 8; ++a) {
      float mx = fmaxf(S[0].f[a], S[1].f[a]);
      mx = fmaxf(mx, __shfl_xor(mx, 1, 32));
      mx = fmaxf(mx, __shfl_xor(mx, 2, 32));
      mx = fmaxf(mx, __shfl_xor(mx, 4, 32));
      mx = fmaxf(mx, __shfl_xor(mx, 8, 32));
      const float mnew  = fmaxf(mi[a], mx);
      const float alpha = __expf(mi[a] - mnew);
      const float p0 = __expf(S[0].f[a] - mnew);
      const float p1 = __expf(S[1].f[a] - mnew);
      float rs = p0 + p1;
      rs += __shfl_xor(rs, 1, 32);
      rs += __shfl_xor(rs, 2, 32);
      rs += __shfl_xor(rs, 4, 32);
      rs += __shfl_xor(rs, 8, 32);
      li[a] = li[a] * alpha + rs;
      mi[a] = mnew;
#pragma unroll
      for (int nt = 0; nt < 4; ++nt) O[nt].f[a] *= alpha;
      const int m = a + half * 8;
      myP[m * 32 + lan]      = to_bf16(p0);
      myP[m * 32 + 16 + lan] = to_bf16(p1);
    }
    asm volatile("s_wait_dscnt 0x0" ::: "memory");  // wave-level P visibility
    // --- P (16x32) A-fragment from LDS ---
    ABFrag pfr;
    {
      const __bf16* prow = &myP[lan * 32];
      pfr.q[0] = *(const U4*)(prow + half * 16);
      pfr.q[1] = *(const U4*)(prow + half * 16 + 8);
    }
    // --- O += P @ V  (V stored transposed (b,h,dk,t): contiguous B-frags) ---
#pragma unroll
    for (int nt = 0; nt < 4; ++nt) {
      ABFrag vf;
      const __bf16* vrow =
          vT + ((size_t)((b * H + h) * DK + nt * 16 + lan)) * T + j0;
      vf.q[0] = *(const U4*)(vrow + half * 16);
      vf.q[1] = *(const U4*)(vrow + half * 16 + 8);
      O[nt].v = wmma_bf16(pfr.v, vf.v, O[nt].v);
    }
  }
  // --- normalize and write x (B,T,F) fp32 ---
#pragma unroll
  for (int a = 0; a < 8; ++a) {
    const float inv = 1.0f / li[a];
    const int m = a + half * 8;
    float* orow = xout + ((size_t)(b * T + i0 + m)) * F + h * 64;
#pragma unroll
    for (int nt = 0; nt < 4; ++nt) orow[nt * 16 + lan] = O[nt].f[a] * inv;
  }
}

// ---------------------------------------------------------------------------
// Launcher
// ---------------------------------------------------------------------------
extern "C" void kernel_launch(void* const* d_in, const int* in_sizes, int n_in,
                              void* d_out, int out_size, void* d_ws,
                              size_t ws_size, hipStream_t stream) {
  const float* query = (const float*)d_in[0];
  const float* key   = (const float*)d_in[1];
  const float* value = (const float*)d_in[2];
  const float* pos   = (const float*)d_in[3];
  const int*   mask  = (const int*)d_in[4];
  const float* Wq = (const float*)d_in[5];
  const float* bq = (const float*)d_in[6];
  const float* Wk = (const float*)d_in[7];
  const float* bk = (const float*)d_in[8];
  const float* Wv = (const float*)d_in[9];
  const float* bv = (const float*)d_in[10];
  const float* Wpos = (const float*)d_in[11];
  const float* Wo = (const float*)d_in[12];
  const float* bo = (const float*)d_in[13];
  const float* pu = (const float*)d_in[14];  // pos_bias_u (flattened F)
  const float* pv = (const float*)d_in[15];  // pos_bias_v (flattened F)

  // workspace layout (bf16 activations + fp32 context) ~28 MiB
  char* wsp = (char*)d_ws;
  __bf16* quB = (__bf16*)(wsp + (size_t)(0u  << 20));
  __bf16* qvB = (__bf16*)(wsp + (size_t)(4u  << 20));
  __bf16* kB  = (__bf16*)(wsp + (size_t)(8u  << 20));
  __bf16* vTB = (__bf16*)(wsp + (size_t)(12u << 20));
  __bf16* pB  = (__bf16*)(wsp + (size_t)(16u << 20));
  float*  xB  = (float*)(wsp + (size_t)(20u << 20));

  const int Fv = 1024, Tv = 1024, Bv = 2, Pv = 2047;
  dim3 blk(256);
  dim3 gq((Bv * Tv + 127) / 128, Fv / 64);
  // q -> q_u, q_v (column-broadcast pos biases fused in epilogue)
  gemm_bf16_wmma<<<gq, blk, 0, stream>>>(query, Wq, bq, pu, pv, quB, qvB,
                                         Bv * Tv, Fv, Fv, 0, Tv);
  gemm_bf16_wmma<<<gq, blk, 0, stream>>>(key, Wk, bk, nullptr, nullptr, kB,
                                         nullptr, Bv * Tv, Fv, Fv, 0, Tv);
  gemm_bf16_wmma<<<gq, blk, 0, stream>>>(value, Wv, bv, nullptr, nullptr, vTB,
                                         nullptr, Bv * Tv, Fv, Fv, 2, Tv);
  dim3 gp((Pv + 127) / 128, Fv / 64);
  gemm_bf16_wmma<<<gp, blk, 0, stream>>>(pos, Wpos, nullptr, nullptr, nullptr,
                                         pB, nullptr, Pv, Fv, Fv, 0, Tv);
  // fused rel-pos flash attention: 2048 waves -> 256 blocks x 8 waves
  relattn_wmma<<<dim3(256), blk, 0, stream>>>(quB, qvB, kB, vTB, pB, mask, xB);
  // output projection (fp32 out)
  gemm_bf16_wmma<<<gq, blk, 0, stream>>>(xB, Wo, bo, nullptr, nullptr, d_out,
                                         nullptr, Bv * Tv, Fv, Fv, 1, Tv);
}
